// Transformer_38886633898765
// MI455X (gfx1250) — compile-verified, matrix operand order checked by reading, not run on hardware
//
#include <hip/hip_runtime.h>

// ---------------------------------------------------------------------------
// Transformer encoder-decoder for MI455X (gfx1250, wave32, WMMA).
// Dense math: V_WMMA_F32_16X16X32_BF16 (bf16 in, f32 accum).
// Tile staging: GLOBAL_LOAD_ASYNC_TO_LDS_B64 + double-buffered LDS.
// ---------------------------------------------------------------------------

#define SEQL 1024
#define DM   512
#define NH   8
#define HDM  4096   // NH * DM
#define FFD  2048   // 4 * DM

typedef __attribute__((ext_vector_type(16))) __bf16 v16bf;
typedef __attribute__((ext_vector_type(8)))  float  v8f;

__device__ __forceinline__ unsigned short f2bf(float f) {
  unsigned int u = __float_as_uint(f);
  u += 0x7fffu + ((u >> 16) & 1u);          // round-to-nearest-even
  return (unsigned short)(u >> 16);
}

// Async global->LDS copy of 8 bytes (CDNA5 TDM-adjacent async path, ASYNCcnt).
// LDS address = low 32 bits of the generic pointer.
__device__ __forceinline__ void async_g2l_b64(const unsigned short* g,
                                              unsigned short* l) {
  unsigned lds = (unsigned)(unsigned long long)l;
  asm volatile("global_load_async_to_lds_b64 %0, %1, off"
               :: "v"(lds), "v"((unsigned long long)g) : "memory");
}
__device__ __forceinline__ void wait_async0() {
  asm volatile("s_wait_asynccnt 0" ::: "memory");
}

// ---------------------------------------------------------------------------
// GEMM: C[M,N] = alpha * A[M,K] * B + bias (+ReLU), A/B bf16, accum f32.
// TRANSB=false: B is [K,N] row-major.  TRANSB=true: B is [N,K] row-major.
// Block tile 128xBN (BN = 64 or 128), BK=32; 8 waves, each wave owns a
// 32x(BN/2) sub-tile via 2x(BN/32) WMMA 16x16x32 tiles.
// Double-buffered LDS; async engine stages tile k+1 while tile k computes.
// Batched via blockIdx.z with element strides Ab/Bb/Cb/Cbfb (for the 8 heads).
// Requires M%128==0, N%BN==0, K%32==0 (true for every call in this model).
// ---------------------------------------------------------------------------
template<bool TRANSB, int BN>
__global__ __launch_bounds__(256)
void gemm_bf16_wmma(const unsigned short* __restrict__ A, int lda, long long Ab,
                    const unsigned short* __restrict__ B, int ldb, long long Bb,
                    float* __restrict__ C, long long Cb,
                    unsigned short* __restrict__ Cbf, long long Cbfb,
                    int ldc, const float* __restrict__ bias,
                    float alpha, int relu, int K)
{
  constexpr int BM = 128, BK = 32, LS = BK + 8;   // LDS row stride (pad)
  constexpr int JT = BN / 32;                     // B tiles per wave (2 or 4)
  constexpr int BITERS = BN / 32;                 // B staging iterations
  __shared__ unsigned short As[2][BM * LS];
  __shared__ unsigned short Bs[2][BN * LS];       // stored as [n][k]

  const int bz = blockIdx.z;
  A += (long long)bz * Ab;
  B += (long long)bz * Bb;
  if (C)   C   += (long long)bz * Cb;
  if (Cbf) Cbf += (long long)bz * Cbfb;

  const int m0 = blockIdx.y * BM, n0 = blockIdx.x * BN;
  const int tid = threadIdx.x, lane = tid & 31, wave = tid >> 5;
  const int wm0 = (wave & 3) * 32, wn0 = (wave >> 2) * (16 * JT);
  const int l16 = lane & 15, koff = (lane >> 4) * 8;       // wave32 WMMA layout

  v8f acc[2][JT] = {};
  union F { v16bf v; uint4 q[2]; };

  // ---- stage tile kb into LDS buffer buf (async where possible) -----------
  auto stage = [&](int kb, int buf) {
    #pragma unroll
    for (int i = 0; i < 4; ++i) {              // A: 128x32, 8B per lane-slot
      int v = tid + i * 256;
      int r = v >> 3, kc = (v & 7) * 4;
      async_g2l_b64(A + (size_t)(m0 + r) * lda + kb + kc, &As[buf][r * LS + kc]);
    }
    if (TRANSB) {                              // B rows are K-contiguous
      #pragma unroll
      for (int i = 0; i < BITERS; ++i) {
        int v = tid + i * 256;
        int r = v >> 3, kc = (v & 7) * 4;      // r = n, kc = k
        async_g2l_b64(B + (size_t)(n0 + r) * ldb + kb + kc, &Bs[buf][r * LS + kc]);
      }
    } else {                                   // transpose during staging
      constexpr int NV = BN / 4;               // vec4 slots per k-row
      #pragma unroll
      for (int i = 0; i < BITERS; ++i) {
        int v = tid + i * 256;
        int k = v / NV, nc = (v % NV) * 4;
        uint2 d = *reinterpret_cast<const uint2*>(B + (size_t)(kb + k) * ldb + n0 + nc);
        Bs[buf][(nc + 0) * LS + k] = (unsigned short)(d.x & 0xffffu);
        Bs[buf][(nc + 1) * LS + k] = (unsigned short)(d.x >> 16);
        Bs[buf][(nc + 2) * LS + k] = (unsigned short)(d.y & 0xffffu);
        Bs[buf][(nc + 3) * LS + k] = (unsigned short)(d.y >> 16);
      }
    }
  };

  // ---- fragments + 2xJT WMMA from LDS buffer buf --------------------------
  auto compute = [&](int buf) {
    F af[2], bfm[JT];
    #pragma unroll
    for (int t = 0; t < 2; ++t) {
      int r = wm0 + t * 16 + l16;              // A row
      af[t].q[0] = *reinterpret_cast<const uint4*>(&As[buf][r * LS + koff]);
      af[t].q[1] = *reinterpret_cast<const uint4*>(&As[buf][r * LS + 16 + koff]);
    }
    #pragma unroll
    for (int j = 0; j < JT; ++j) {
      int c = wn0 + j * 16 + l16;              // B col
      bfm[j].q[0] = *reinterpret_cast<const uint4*>(&Bs[buf][c * LS + koff]);
      bfm[j].q[1] = *reinterpret_cast<const uint4*>(&Bs[buf][c * LS + 16 + koff]);
    }
    #pragma unroll
    for (int i = 0; i < 2; ++i)
      #pragma unroll
      for (int j = 0; j < JT; ++j)
        acc[i][j] = __builtin_amdgcn_wmma_f32_16x16x32_bf16(
            false, af[i].v, false, bfm[j].v, (short)0, acc[i][j], false, false);
  };

  // ---- double-buffered main loop ------------------------------------------
  stage(0, 0);
  wait_async0();
  __syncthreads();
  int p = 0;
  for (int kb = 0; kb < K; kb += BK) {
    if (kb + BK < K) stage(kb + BK, 1 - p);    // async fill of next buffer
    compute(p);                                // WMMAs on current buffer
    wait_async0();                             // own async transfers landed
    __syncthreads();                           // all waves done reading buf p
    p ^= 1;
  }

  // ---- epilogue: alpha, bias, relu, f32 and/or bf16 stores ----------------
  const int lh = lane >> 4;
  #pragma unroll
  for (int i = 0; i < 2; ++i) {
    #pragma unroll
    for (int j = 0; j < JT; ++j) {
      int col = n0 + wn0 + j * 16 + l16;
      float bv = bias ? bias[col] : 0.0f;
      #pragma unroll
      for (int r = 0; r < 8; ++r) {
        int row = m0 + wm0 + i * 16 + 8 * lh + r;
        float vv = alpha * acc[i][j][r] + bv;
        if (relu) vv = fmaxf(vv, 0.0f);
        if (C)   C[(size_t)row * ldc + col] = vv;
        if (Cbf) Cbf[(size_t)row * ldc + col] = f2bf(vv);
      }
    }
  }
}

// ---------------------------------------------------------------------------
// Row softmax over [NH, SEQL, SEQL] f32 scores -> bf16 probs; causal optional.
// ---------------------------------------------------------------------------
__global__ __launch_bounds__(256)
void softmax_bf16(const float* __restrict__ S, unsigned short* __restrict__ P,
                  int causal)
{
  const int row = blockIdx.x, head = blockIdx.y, tid = threadIdx.x;
  const size_t base = ((size_t)head * SEQL + row) * SEQL;
  const int limit = causal ? (row + 1) : SEQL;
  __shared__ float red[256];

  float mx = -3.4e38f;
  #pragma unroll
  for (int i = 0; i < SEQL / 256; ++i) {
    int c = tid + i * 256;
    if (c < limit) mx = fmaxf(mx, S[base + c]);
  }
  red[tid] = mx; __syncthreads();
  for (int o = 128; o > 0; o >>= 1) {
    if (tid < o) red[tid] = fmaxf(red[tid], red[tid + o]);
    __syncthreads();
  }
  mx = red[0]; __syncthreads();

  float ev[SEQL / 256]; float sum = 0.f;
  #pragma unroll
  for (int i = 0; i < SEQL / 256; ++i) {
    int c = tid + i * 256;
    float e = (c < limit) ? __expf(S[base + c] - mx) : 0.f;
    ev[i] = e; sum += e;
  }
  red[tid] = sum; __syncthreads();
  for (int o = 128; o > 0; o >>= 1) {
    if (tid < o) red[tid] += red[tid + o];
    __syncthreads();
  }
  const float inv = 1.f / red[0];
  #pragma unroll
  for (int i = 0; i < SEQL / 256; ++i)
    P[base + tid + i * 256] = f2bf(ev[i] * inv);
}

// ---------------------------------------------------------------------------
// y = LayerNorm(x + r) * g + b ; writes f32 residual stream + bf16 copy.
// ---------------------------------------------------------------------------
__global__ __launch_bounds__(128)
void add_layernorm(const float* __restrict__ X, const float* __restrict__ R,
                   const float* __restrict__ gw, const float* __restrict__ bw,
                   float* __restrict__ Yf, unsigned short* __restrict__ Ybf)
{
  const int row = blockIdx.x, tid = threadIdx.x;
  const size_t base = (size_t)row * DM;
  __shared__ float red[128];

  float v[DM / 128]; float s = 0.f, q = 0.f;
  #pragma unroll
  for (int i = 0; i < DM / 128; ++i) {
    int c = tid + i * 128;
    float t = X[base + c] + R[base + c];
    v[i] = t; s += t; q += t * t;
  }
  red[tid] = s; __syncthreads();
  for (int o = 64; o > 0; o >>= 1) { if (tid < o) red[tid] += red[tid + o]; __syncthreads(); }
  s = red[0]; __syncthreads();
  red[tid] = q; __syncthreads();
  for (int o = 64; o > 0; o >>= 1) { if (tid < o) red[tid] += red[tid + o]; __syncthreads(); }
  q = red[0];

  const float mu   = s * (1.f / DM);
  const float var  = q * (1.f / DM) - mu * mu;
  const float rstd = rsqrtf(var + 1e-5f);
  #pragma unroll
  for (int i = 0; i < DM / 128; ++i) {
    int c = tid + i * 128;
    float y = (v[i] - mu) * rstd * gw[c] + bw[c];
    Yf[base + c]  = y;
    Ybf[base + c] = f2bf(y);
  }
}

// Pack per-head weights [NH,512,512] f32 -> [512, NH*512] bf16 (all heads as
// one GEMM B matrix: out[d, h*512 + e] = W[h, d, e]).
__global__ __launch_bounds__(256)
void pack_heads_bf16(const float* __restrict__ W, unsigned short* __restrict__ O)
{
  unsigned idx = blockIdx.x * 256u + threadIdx.x;        // NH*512*512 = 2^21
  unsigned e = idx & 511u, d = (idx >> 9) & 511u, h = idx >> 18;
  O[(size_t)d * HDM + h * DM + e] = f2bf(W[idx]);
}

__global__ __launch_bounds__(256)
void convert_bf16_k(const float* __restrict__ X, unsigned short* __restrict__ O)
{
  size_t idx = (size_t)blockIdx.x * 256u + threadIdx.x;  // n % 256 == 0
  O[idx] = f2bf(X[idx]);
}

// ---------------------------------------------------------------------------
// Host-side orchestration
// ---------------------------------------------------------------------------
static const size_t OFF_XF  = 0;                                   // f32 [1024,512]
static const size_t OFF_XBF = OFF_XF  + (size_t)SEQL * DM * 4;     // bf16 copy
static const size_t OFF_ENC = OFF_XBF + (size_t)SEQL * DM * 2;     // enc out bf16
static const size_t OFF_WP  = OFF_ENC + (size_t)SEQL * DM * 2;     // packed weight bf16 (<=2M elems)
static const size_t OFF_QA  = OFF_WP  + (size_t)DM * HDM * 2;      // [1024,4096] bf16
static const size_t OFF_KA  = OFF_QA  + (size_t)SEQL * HDM * 2;
static const size_t OFF_VA  = OFF_KA  + (size_t)SEQL * HDM * 2;
static const size_t OFF_CAT = OFF_VA  + (size_t)SEQL * HDM * 2;    // head concat bf16
static const size_t OFF_SC  = OFF_CAT + (size_t)SEQL * HDM * 2;    // scores f32 [8,1024,1024]
static const size_t OFF_PR  = OFF_SC  + (size_t)NH * SEQL * SEQL * 4; // probs bf16
static const size_t OFF_FFH = OFF_PR  + (size_t)NH * SEQL * SEQL * 2; // ff hidden bf16
static const size_t OFF_TMP = OFF_FFH + (size_t)SEQL * FFD * 2;    // f32 [1024,512]

static inline char* wsp(void* ws, size_t off) { return (char*)ws + off; }

struct MHAP { const float *wq, *bq, *wk, *bk, *wv, *bv, *wo, *bo; };
struct FFP  { const float *b1, *b2, *w1, *w2; };
struct LNP  { const float *b, *g; };

// jax pytree leaf order within an MHA dict (sorted keys): bk,bo,bq,bv,wk,wo,wq,wv
static MHAP mhap(void* const* d, int b) {
  MHAP p;
  p.bk = (const float*)d[b + 0]; p.bo = (const float*)d[b + 1];
  p.bq = (const float*)d[b + 2]; p.bv = (const float*)d[b + 3];
  p.wk = (const float*)d[b + 4]; p.wo = (const float*)d[b + 5];
  p.wq = (const float*)d[b + 6]; p.wv = (const float*)d[b + 7];
  return p;
}

static void launch_gemm(hipStream_t st, bool transB,
                        const void* A, int lda, long long Ab,
                        const void* B, int ldb, long long Bb,
                        float* C, long long Cb,
                        void* Cbf, long long Cbfb, int ldc,
                        const float* bias, float alpha, int relu,
                        int M, int N, int K, int batch)
{
  const unsigned short* As = (const unsigned short*)A;
  const unsigned short* Bs = (const unsigned short*)B;
  unsigned short* Cb16 = (unsigned short*)Cbf;
  dim3 blk(256, 1, 1);
  if ((N % 128) == 0) {
    dim3 g(N / 128, M / 128, batch);
    if (transB)
      gemm_bf16_wmma<true, 128><<<g, blk, 0, st>>>(As, lda, Ab, Bs, ldb, Bb,
          C, Cb, Cb16, Cbfb, ldc, bias, alpha, relu, K);
    else
      gemm_bf16_wmma<false, 128><<<g, blk, 0, st>>>(As, lda, Ab, Bs, ldb, Bb,
          C, Cb, Cb16, Cbfb, ldc, bias, alpha, relu, K);
  } else {
    dim3 g(N / 64, M / 128, batch);
    if (transB)
      gemm_bf16_wmma<true, 64><<<g, blk, 0, st>>>(As, lda, Ab, Bs, ldb, Bb,
          C, Cb, Cb16, Cbfb, ldc, bias, alpha, relu, K);
    else
      gemm_bf16_wmma<false, 64><<<g, blk, 0, st>>>(As, lda, Ab, Bs, ldb, Bb,
          C, Cb, Cb16, Cbfb, ldc, bias, alpha, relu, K);
  }
}

// Full multi-head attention: out (pre-LN, f32) -> OFF_TMP
static void run_mha(hipStream_t st, void* ws, const void* q_bf, const void* kv_bf,
                    const MHAP& p, int causal)
{
  unsigned short* WP  = (unsigned short*)wsp(ws, OFF_WP);
  unsigned short* QA  = (unsigned short*)wsp(ws, OFF_QA);
  unsigned short* KA  = (unsigned short*)wsp(ws, OFF_KA);
  unsigned short* VA  = (unsigned short*)wsp(ws, OFF_VA);
  unsigned short* CAT = (unsigned short*)wsp(ws, OFF_CAT);
  unsigned short* PR  = (unsigned short*)wsp(ws, OFF_PR);
  float* SC  = (float*)wsp(ws, OFF_SC);
  float* TMP = (float*)wsp(ws, OFF_TMP);

  const int PK = (NH * DM * DM) / 256;   // pack grid

  // Q/K/V projections for all 8 heads as single [1024,512]x[512,4096] GEMMs
  pack_heads_bf16<<<PK, 256, 0, st>>>(p.wq, WP);
  launch_gemm(st, false, q_bf, DM, 0, WP, HDM, 0, nullptr, 0, QA, 0, HDM,
              p.bq, 1.f, 0, SEQL, HDM, DM, 1);
  pack_heads_bf16<<<PK, 256, 0, st>>>(p.wk, WP);
  launch_gemm(st, false, kv_bf, DM, 0, WP, HDM, 0, nullptr, 0, KA, 0, HDM,
              p.bk, 1.f, 0, SEQL, HDM, DM, 1);
  pack_heads_bf16<<<PK, 256, 0, st>>>(p.wv, WP);
  launch_gemm(st, false, kv_bf, DM, 0, WP, HDM, 0, nullptr, 0, VA, 0, HDM,
              p.bv, 1.f, 0, SEQL, HDM, DM, 1);

  // scores[h] = (Q_h @ K_h^T) / sqrt(64), batched over heads (blockIdx.z)
  launch_gemm(st, true, QA, HDM, DM, KA, HDM, DM,
              SC, (long long)SEQL * SEQL, nullptr, 0, SEQL,
              nullptr, 0.125f, 0, SEQL, SEQL, DM, NH);

  softmax_bf16<<<dim3(SEQL, NH), 256, 0, st>>>(SC, PR, causal);

  // O_h = P_h @ V_h, written directly into concat layout [1024, 4096]
  launch_gemm(st, false, PR, SEQL, (long long)SEQL * SEQL, VA, HDM, DM,
              nullptr, 0, CAT, DM, HDM, nullptr, 1.f, 0, SEQL, DM, SEQL, NH);

  // output projection: concat [1024,4096] @ wo [4096,512] + bo
  convert_bf16_k<<<(HDM * DM) / 256, 256, 0, st>>>(p.wo, WP);
  launch_gemm(st, false, CAT, HDM, 0, WP, DM, 0, TMP, 0, nullptr, 0, DM,
              p.bo, 1.f, 0, SEQL, DM, HDM, 1);
}

static void run_ff(hipStream_t st, void* ws, const void* x_bf, const FFP& f)
{
  unsigned short* WP  = (unsigned short*)wsp(ws, OFF_WP);
  unsigned short* FFH = (unsigned short*)wsp(ws, OFF_FFH);
  float* TMP = (float*)wsp(ws, OFF_TMP);

  convert_bf16_k<<<(DM * FFD) / 256, 256, 0, st>>>(f.w1, WP);
  launch_gemm(st, false, x_bf, DM, 0, WP, FFD, 0, nullptr, 0, FFH, 0, FFD,
              f.b1, 1.f, 1 /*relu*/, SEQL, FFD, DM, 1);
  convert_bf16_k<<<(FFD * DM) / 256, 256, 0, st>>>(f.w2, WP);
  launch_gemm(st, false, FFH, FFD, 0, WP, DM, 0, TMP, 0, nullptr, 0, DM,
              f.b2, 1.f, 0, SEQL, DM, FFD, 1);
}

extern "C" void kernel_launch(void* const* d_in, const int* in_sizes, int n_in,
                              void* d_out, int out_size, void* d_ws, size_t ws_size,
                              hipStream_t stream)
{
  (void)in_sizes; (void)n_in; (void)out_size; (void)ws_size;
  // Input layout (jax pytree leaf order, dict keys sorted at every level):
  //   0: x_enc, 1: x_dec,
  //   2 + 26*L       : decoder layer L -> ca(8), ff(4), ln1(2), ln2(2), ln3(2), sa(8)
  //   2 + 156 + 16*L : encoder layer L -> ff(4), ln1(2), ln2(2), mha(8)
  //   then mask_causal / mask_pad_dec / mask_pad_ed (pads are zero; causal is
  //   applied analytically in the softmax kernel).
  float*          XF  = (float*)wsp(d_ws, OFF_XF);
  unsigned short* XBF = (unsigned short*)wsp(d_ws, OFF_XBF);
  unsigned short* ENC = (unsigned short*)wsp(d_ws, OFF_ENC);
  float*          TMP = (float*)wsp(d_ws, OFF_TMP);
  const int CVX = (SEQL * DM) / 256;

  // ---------------- encoder ----------------
  hipMemcpyAsync(XF, d_in[0], (size_t)SEQL * DM * 4, hipMemcpyDeviceToDevice, stream);
  convert_bf16_k<<<CVX, 256, 0, stream>>>((const float*)d_in[0], XBF);
  for (int L = 0; L < 6; ++L) {
    const int b = 2 + 156 + 16 * L;
    FFP  ff  = { (const float*)d_in[b+0], (const float*)d_in[b+1],
                 (const float*)d_in[b+2], (const float*)d_in[b+3] };
    LNP  ln1 = { (const float*)d_in[b+4], (const float*)d_in[b+5] };
    LNP  ln2 = { (const float*)d_in[b+6], (const float*)d_in[b+7] };
    MHAP m   = mhap(d_in, b + 8);

    run_mha(stream, d_ws, XBF, XBF, m, 0);
    add_layernorm<<<SEQL, 128, 0, stream>>>(XF, TMP, ln1.g, ln1.b, XF, XBF);
    run_ff(stream, d_ws, XBF, ff);
    add_layernorm<<<SEQL, 128, 0, stream>>>(XF, TMP, ln2.g, ln2.b, XF, XBF);
  }
  hipMemcpyAsync(ENC, XBF, (size_t)SEQL * DM * 2, hipMemcpyDeviceToDevice, stream);

  // ---------------- decoder ----------------
  hipMemcpyAsync(XF, d_in[1], (size_t)SEQL * DM * 4, hipMemcpyDeviceToDevice, stream);
  convert_bf16_k<<<CVX, 256, 0, stream>>>((const float*)d_in[1], XBF);
  for (int L = 0; L < 6; ++L) {
    const int b = 2 + 26 * L;
    MHAP ca  = mhap(d_in, b + 0);
    FFP  ff  = { (const float*)d_in[b+8],  (const float*)d_in[b+9],
                 (const float*)d_in[b+10], (const float*)d_in[b+11] };
    LNP  ln1 = { (const float*)d_in[b+12], (const float*)d_in[b+13] };
    LNP  ln2 = { (const float*)d_in[b+14], (const float*)d_in[b+15] };
    LNP  ln3 = { (const float*)d_in[b+16], (const float*)d_in[b+17] };
    MHAP sa  = mhap(d_in, b + 18);

    run_mha(stream, d_ws, XBF, XBF, sa, 1);                 // self-attn, causal
    add_layernorm<<<SEQL, 128, 0, stream>>>(XF, TMP, ln1.g, ln1.b, XF, XBF);
    run_mha(stream, d_ws, XBF, ENC, ca, 1);                 // cross-attn (ref applies causal too)
    add_layernorm<<<SEQL, 128, 0, stream>>>(XF, TMP, ln2.g, ln2.b, XF, XBF);
    run_ff(stream, d_ws, XBF, ff);
    add_layernorm<<<SEQL, 128, 0, stream>>>(XF, TMP, ln3.g, ln3.b, XF, XBF);
  }

  hipMemcpyAsync(d_out, XF, (size_t)SEQL * DM * 4, hipMemcpyDeviceToDevice, stream);
}
